// LowRankAttention_69303592288307
// MI455X (gfx1250) — matrix operand order repaired and checked by reading, not compiled, not run on hardware
//
#include <hip/hip_runtime.h>
#include <hip/hip_bf16.h>

typedef __attribute__((ext_vector_type(16))) __bf16 v16bf;
typedef __attribute__((ext_vector_type(8)))  __bf16 v8bf;
typedef __attribute__((ext_vector_type(4)))  __bf16 v4bf;
typedef __attribute__((ext_vector_type(8)))  float  v8f;
typedef unsigned int v4u __attribute__((ext_vector_type(4)));
typedef int          v8i __attribute__((ext_vector_type(8)));
typedef int          v4i __attribute__((ext_vector_type(4)));

#if __has_builtin(__builtin_amdgcn_tensor_load_to_lds) && \
    __has_builtin(__builtin_amdgcn_s_wait_tensorcnt)
#define HAVE_TDM 1
#endif

static __device__ __forceinline__ v16bf frag_cat(v8bf lo, v8bf hi) {
  v16bf r;
#pragma unroll
  for (int i = 0; i < 8; ++i) { r[i] = lo[i]; r[i + 8] = hi[i]; }
  return r;
}

static __device__ __forceinline__ v8f wmma_bf16(v16bf a, v16bf b, v8f c) {
  // (neg_a, A, neg_b, B, c_mod, C, reuse_a, reuse_b)
  return __builtin_amdgcn_wmma_f32_16x16x32_bf16(false, a, false, b, (short)0, c,
                                                 false, false);
}

// Convert 8 fp32 (two float4) -> 8 bf16 stored contiguously (16B ds_store)
static __device__ __forceinline__ void cvt_store8(__bf16* dst, float4 a, float4 b) {
  v8bf v;
  v[0] = (__bf16)a.x; v[1] = (__bf16)a.y; v[2] = (__bf16)a.z; v[3] = (__bf16)a.w;
  v[4] = (__bf16)b.x; v[5] = (__bf16)b.y; v[6] = (__bf16)b.z; v[7] = (__bf16)b.w;
  *(v8bf*)dst = v;
}

// A fragment: 16x32 bf16, LDS tile row-major [M][32]; lanes 0-15 rows, lane>=16 K+8
static __device__ __forceinline__ v16bf load_afrag(const __bf16* tile, int m0, int lane) {
  const int l = lane & 15, h = (lane >> 4) << 3;   // K base 0 or 8
  const __bf16* p = tile + (m0 + l) * 32 + h;
  return frag_cat(*(const v8bf*)p, *(const v8bf*)(p + 16)); // K{h..h+7}, K{h+16..h+23}
}

// B fragment: 32x16 bf16 with LDS tile stored as [N][32K]; lanes>=16 K+8 (same order as A)
static __device__ __forceinline__ v16bf load_bfrag(const __bf16* tile, int n0, int lane) {
  const int l = lane & 15, h = (lane >> 4) << 3;   // K base 0 or 8
  const __bf16* p = tile + (n0 + l) * 32 + h;
  return frag_cat(*(const v8bf*)p, *(const v8bf*)(p + 16));
}

static __device__ __forceinline__ v8f vzero8() {
  v8f z = {0.f, 0.f, 0.f, 0.f, 0.f, 0.f, 0.f, 0.f};
  return z;
}

#ifdef HAVE_TDM
// LDS byte offset of a __shared__ object (generic -> AS(3) -> integer)
static __device__ __forceinline__ unsigned lds_offset(const void* p) {
  return (unsigned)(unsigned long long)(const __attribute__((address_space(3))) void*)p;
}

// Issue a TDM 2D tile load: tile_w (contiguous, elems) x tile_h rows of bf16,
// global row stride row_stride (elems), packed contiguously into LDS at lds_b.
static __device__ __forceinline__ void tdm_load_tile_bf16(
    unsigned lds_b, const __bf16* gptr, int tile_w, int tile_h, int row_stride) {
  const unsigned long long ga = (unsigned long long)gptr;
  v4u g0;
  g0[0] = 1u;                                     // count=1, no gather
  g0[1] = lds_b;                                  // lds_addr
  g0[2] = (unsigned)ga;                           // global_addr[31:0]
  g0[3] = (unsigned)(ga >> 32) | 0x80000000u;     // global_addr[56:32] | type=2
  v8i g1;
  g1[0] = 1 << 16;                                // data_size = 1 (2 bytes)
  g1[1] = (tile_w & 0xffff) << 16;                // tensor_dim0[15:0]
  g1[2] = ((unsigned)tile_w >> 16) | ((tile_h & 0xffff) << 16);  // dim0 hi | dim1 lo
  g1[3] = ((unsigned)tile_h >> 16) | ((tile_w & 0xffff) << 16);  // dim1 hi | tile_dim0
  g1[4] = tile_h & 0xffff;                        // tile_dim1 (tile_dim2 = 0)
  g1[5] = row_stride;                             // tensor_dim0_stride[31:0]
  g1[6] = 0;                                      // stride hi | dim1_stride lo (2D)
  g1[7] = 0;
  v4i gz4 = {0, 0, 0, 0};
  v8i gz8 = {0, 0, 0, 0, 0, 0, 0, 0};
  // 6-arg toolchain variant: (g0, g1, g2, g3, g4, cpol)
  __builtin_amdgcn_tensor_load_to_lds(g0, g1, gz4, gz4, gz8, 0);
}
#endif

// -------- Kernel 1: kv = x @ w_kv^T  (M=16384, N=2048, K=1024), out bf16 --------
__global__ __launch_bounds__(256) void kv_gemm_kernel(
    const float* __restrict__ X, const float* __restrict__ W, __bf16* __restrict__ KV) {
  __shared__ __align__(16) __bf16 As[2][128 * 32];
  __shared__ __align__(16) __bf16 Bs[2][128 * 32];
  const int tid = threadIdx.x, lane = tid & 31, wave = tid >> 5;
  const int mb = blockIdx.y * 128, nb = blockIdx.x * 128;
  const int wm = (wave & 3) * 32, wn = (wave >> 2) * 64;
  v8f acc[2][4];
#pragma unroll
  for (int i = 0; i < 2; ++i)
#pragma unroll
    for (int j = 0; j < 4; ++j) acc[i][j] = vzero8();

  const int lr = tid >> 1, lk = (tid & 1) * 16;  // 128 rows x 2 halves of 16 floats
  const float* xrow = X + (size_t)(mb + lr) * 1024 + lk;
  const float* wrow = W + (size_t)(nb + lr) * 1024 + lk;

  {  // prologue: stage tile k0=0 into buffer 0
    const float4* ap = (const float4*)xrow;
    const float4* bp = (const float4*)wrow;
    cvt_store8(&As[0][lr * 32 + lk], ap[0], ap[1]);
    cvt_store8(&As[0][lr * 32 + lk + 8], ap[2], ap[3]);
    cvt_store8(&Bs[0][lr * 32 + lk], bp[0], bp[1]);
    cvt_store8(&Bs[0][lr * 32 + lk + 8], bp[2], bp[3]);
  }
  __syncthreads();

  int cur = 0;
  for (int k0 = 0; k0 < 1024; k0 += 32) {
    const bool nxt = (k0 + 32) < 1024;
    float4 a0, a1, a2, a3, b0, b1, b2, b3;
    if (nxt) {  // issue next tile's global loads before the WMMA burst
      const float4* ap = (const float4*)(xrow + k0 + 32);
      const float4* bp = (const float4*)(wrow + k0 + 32);
      a0 = ap[0]; a1 = ap[1]; a2 = ap[2]; a3 = ap[3];
      b0 = bp[0]; b1 = bp[1]; b2 = bp[2]; b3 = bp[3];
    }
    v16bf af[2], bfr[4];
#pragma unroll
    for (int mi = 0; mi < 2; ++mi) af[mi] = load_afrag(As[cur], wm + mi * 16, lane);
#pragma unroll
    for (int ni = 0; ni < 4; ++ni) bfr[ni] = load_bfrag(Bs[cur], wn + ni * 16, lane);
#pragma unroll
    for (int mi = 0; mi < 2; ++mi)
#pragma unroll
      for (int ni = 0; ni < 4; ++ni) acc[mi][ni] = wmma_bf16(af[mi], bfr[ni], acc[mi][ni]);
    if (nxt) {  // stage into the other buffer while this one is being read
      cvt_store8(&As[cur ^ 1][lr * 32 + lk], a0, a1);
      cvt_store8(&As[cur ^ 1][lr * 32 + lk + 8], a2, a3);
      cvt_store8(&Bs[cur ^ 1][lr * 32 + lk], b0, b1);
      cvt_store8(&Bs[cur ^ 1][lr * 32 + lk + 8], b2, b3);
    }
    __syncthreads();
    cur ^= 1;
  }
  const int l = lane & 15, hi = lane >> 4;
#pragma unroll
  for (int mi = 0; mi < 2; ++mi)
#pragma unroll
    for (int ni = 0; ni < 4; ++ni) {
      const int col = nb + wn + ni * 16 + l;
#pragma unroll
      for (int r = 0; r < 8; ++r) {
        const int row = mb + wm + mi * 16 + hi * 8 + r;
        KV[(size_t)row * 2048 + col] = (__bf16)acc[mi][ni][r];
      }
    }
}

// -------- Kernel 2a: scores = routers[h] @ kv_bh^T * scale  (per (b,h): 64x512, K=256) --------
__global__ __launch_bounds__(512) void score_kernel(
    const float* __restrict__ routers, const __bf16* __restrict__ KV,
    float* __restrict__ attn) {
  __shared__ __align__(16) __bf16 As[64 * 32];
  __shared__ __align__(16) __bf16 Bs[512 * 32];
  const int bh = blockIdx.x, b = bh >> 3, h = bh & 7;
  const int tid = threadIdx.x, lane = tid & 31, wave = tid >> 5;  // 16 waves
  const int wm = (wave >> 3) * 32, wn = (wave & 7) * 64;
  v8f acc[2][4];
#pragma unroll
  for (int i = 0; i < 2; ++i)
#pragma unroll
    for (int j = 0; j < 4; ++j) acc[i][j] = vzero8();

  const float*  R   = routers + (size_t)h * 64 * 256;
  const __bf16* KVb = KV + (size_t)b * 512 * 2048 + h * 256;
  const int ar = tid >> 3, ak = (tid & 7) * 4;  // A: 64 rows x 8x4 floats

  for (int k0 = 0; k0 < 256; k0 += 32) {
#ifdef HAVE_TDM
    // One TDM descriptor pulls the whole 512x32 bf16 kv tile (row stride 2048)
    // into LDS, packed 64B/row -> matches Bs[n*32] layout. Wave 0 issues, all
    // waves see it after s_wait_tensorcnt + workgroup barrier.
    if (wave == 0) {
      tdm_load_tile_bf16(lds_offset(&Bs[0]), KVb + k0, /*w*/32, /*h*/512, /*stride*/2048);
    }
#else
    {
      const v8bf* kp = (const v8bf*)(KVb + (size_t)tid * 2048 + k0);  // kv already bf16
      v8bf t0 = kp[0], t1 = kp[1], t2 = kp[2], t3 = kp[3];
      v8bf* bs = (v8bf*)&Bs[tid * 32];
      bs[0] = t0; bs[1] = t1; bs[2] = t2; bs[3] = t3;
    }
#endif
    float4 a = *(const float4*)(R + (size_t)ar * 256 + k0 + ak);
    v4bf av;
    av[0] = (__bf16)a.x; av[1] = (__bf16)a.y; av[2] = (__bf16)a.z; av[3] = (__bf16)a.w;
    *(v4bf*)&As[ar * 32 + ak] = av;
#ifdef HAVE_TDM
    if (wave == 0) __builtin_amdgcn_s_wait_tensorcnt(0);
#endif
    __syncthreads();
    v16bf af[2], bfr[4];
#pragma unroll
    for (int mi = 0; mi < 2; ++mi) af[mi] = load_afrag(As, wm + mi * 16, lane);
#pragma unroll
    for (int ni = 0; ni < 4; ++ni) bfr[ni] = load_bfrag(Bs, wn + ni * 16, lane);
#pragma unroll
    for (int mi = 0; mi < 2; ++mi)
#pragma unroll
      for (int ni = 0; ni < 4; ++ni) acc[mi][ni] = wmma_bf16(af[mi], bfr[ni], acc[mi][ni]);
    __syncthreads();
  }
  const float qk_scale = 0.0625f;  // 1/sqrt(256)
  float* out = attn + (size_t)bh * 64 * 512;
  const int l = lane & 15, hi = lane >> 4;
#pragma unroll
  for (int mi = 0; mi < 2; ++mi)
#pragma unroll
    for (int ni = 0; ni < 4; ++ni) {
      const int col = wn + ni * 16 + l;
#pragma unroll
      for (int r = 0; r < 8; ++r) {
        const int row = wm + mi * 16 + hi * 8 + r;
        out[(size_t)row * 512 + col] = acc[mi][ni][r] * qk_scale;
      }
    }
}

// -------- Kernel 2b: row softmax over 512, one wave per row, in place --------
__global__ __launch_bounds__(256) void softmax_kernel(float* __restrict__ attn) {
  const int row = blockIdx.x * 8 + (threadIdx.x >> 5);
  const int lane = threadIdx.x & 31;
  float* p = attn + (size_t)row * 512;
  float4 v[4];
  float mx = -3.402823466e38f;
#pragma unroll
  for (int c = 0; c < 4; ++c) {
    v[c] = *(const float4*)(p + lane * 4 + c * 128);
    mx = fmaxf(mx, fmaxf(fmaxf(v[c].x, v[c].y), fmaxf(v[c].z, v[c].w)));
  }
#pragma unroll
  for (int s = 16; s > 0; s >>= 1) mx = fmaxf(mx, __shfl_xor(mx, s, 32));
  float sum = 0.f;
#pragma unroll
  for (int c = 0; c < 4; ++c) {
    v[c].x = __expf(v[c].x - mx); v[c].y = __expf(v[c].y - mx);
    v[c].z = __expf(v[c].z - mx); v[c].w = __expf(v[c].w - mx);
    sum += v[c].x + v[c].y + v[c].z + v[c].w;
  }
#pragma unroll
  for (int s = 16; s > 0; s >>= 1) sum += __shfl_xor(sum, s, 32);
  const float inv = 1.f / sum;
#pragma unroll
  for (int c = 0; c < 4; ++c) {
    v[c].x *= inv; v[c].y *= inv; v[c].z *= inv; v[c].w *= inv;
    *(float4*)(p + lane * 4 + c * 128) = v[c];
  }
}

// -------- Kernel 2c: out_mid = attn @ kv_bh (per (b,h): 64x256, K=512), out bf16 --------
__global__ __launch_bounds__(256) void av_kernel(
    const float* __restrict__ attn, const __bf16* __restrict__ KV,
    __bf16* __restrict__ mid) {
  __shared__ __align__(16) __bf16 As[2][64 * 32];
  __shared__ __align__(16) __bf16 Bs[2][256 * 32];  // transposed: [d][n]
  const int bh = blockIdx.x, b = bh >> 3, h = bh & 7;
  const int tid = threadIdx.x, lane = tid & 31, wave = tid >> 5;  // 8 waves
  const int wm = (wave >> 2) * 32, wn = (wave & 3) * 64;
  v8f acc[2][4];
#pragma unroll
  for (int i = 0; i < 2; ++i)
#pragma unroll
    for (int j = 0; j < 4; ++j) acc[i][j] = vzero8();

  const float*  A   = attn + (size_t)bh * 64 * 512;
  const __bf16* KVb = KV + (size_t)b * 512 * 2048 + h * 256;
  const int ar = tid >> 2, ak = (tid & 3) * 8;   // A: 64 rows x 4x8 floats
  const int bn = tid >> 3, bd = (tid & 7) * 32;  // B: 32 n-rows x 8x32 d

  {  // prologue: tile k0=0 -> buffer 0
    float4 a0 = *(const float4*)(A + (size_t)ar * 512 + ak);
    float4 a1 = *(const float4*)(A + (size_t)ar * 512 + ak + 4);
    cvt_store8(&As[0][ar * 32 + ak], a0, a1);
    const __bf16* kp = KVb + (size_t)bn * 2048 + bd;
    v8bf t0 = ((const v8bf*)kp)[0], t1 = ((const v8bf*)kp)[1];
    v8bf t2 = ((const v8bf*)kp)[2], t3 = ((const v8bf*)kp)[3];
#pragma unroll
    for (int i = 0; i < 8; ++i) {
      Bs[0][(bd + i) * 32 + bn]      = t0[i];
      Bs[0][(bd + 8 + i) * 32 + bn]  = t1[i];
      Bs[0][(bd + 16 + i) * 32 + bn] = t2[i];
      Bs[0][(bd + 24 + i) * 32 + bn] = t3[i];
    }
  }
  __syncthreads();

  int cur = 0;
  for (int k0 = 0; k0 < 512; k0 += 32) {
    const bool nxt = (k0 + 32) < 512;
    float4 a0, a1;
    v8bf t0, t1, t2, t3;
    if (nxt) {
      a0 = *(const float4*)(A + (size_t)ar * 512 + k0 + 32 + ak);
      a1 = *(const float4*)(A + (size_t)ar * 512 + k0 + 32 + ak + 4);
      const __bf16* kp = KVb + (size_t)(k0 + 32 + bn) * 2048 + bd;
      t0 = ((const v8bf*)kp)[0]; t1 = ((const v8bf*)kp)[1];
      t2 = ((const v8bf*)kp)[2]; t3 = ((const v8bf*)kp)[3];
    }
    v16bf af[2], bfr[4];
#pragma unroll
    for (int mi = 0; mi < 2; ++mi) af[mi] = load_afrag(As[cur], wm + mi * 16, lane);
#pragma unroll
    for (int ni = 0; ni < 4; ++ni) bfr[ni] = load_bfrag(Bs[cur], wn + ni * 16, lane);
#pragma unroll
    for (int mi = 0; mi < 2; ++mi)
#pragma unroll
      for (int ni = 0; ni < 4; ++ni) acc[mi][ni] = wmma_bf16(af[mi], bfr[ni], acc[mi][ni]);
    if (nxt) {
      cvt_store8(&As[cur ^ 1][ar * 32 + ak], a0, a1);
#pragma unroll
      for (int i = 0; i < 8; ++i) {
        Bs[cur ^ 1][(bd + i) * 32 + bn]      = t0[i];
        Bs[cur ^ 1][(bd + 8 + i) * 32 + bn]  = t1[i];
        Bs[cur ^ 1][(bd + 16 + i) * 32 + bn] = t2[i];
        Bs[cur ^ 1][(bd + 24 + i) * 32 + bn] = t3[i];
      }
    }
    __syncthreads();
    cur ^= 1;
  }
  const int l = lane & 15, hi = lane >> 4;
#pragma unroll
  for (int mi = 0; mi < 2; ++mi)
#pragma unroll
    for (int ni = 0; ni < 4; ++ni) {
      const int col = wn + ni * 16 + l;
#pragma unroll
      for (int r = 0; r < 8; ++r) {
        const int row = wm + mi * 16 + hi * 8 + r;
        mid[(size_t)(b * 64 + row) * 2048 + h * 256 + col] = (__bf16)acc[mi][ni][r];
      }
    }
}

// -------- Kernel 3: out = mid @ proj_w^T + bias  (M=2048, N=1024, K=2048), fp32 out --------
__global__ __launch_bounds__(256) void proj_kernel(
    const __bf16* __restrict__ Mid, const float* __restrict__ PW,
    const float* __restrict__ bias, float* __restrict__ Out) {
  __shared__ __align__(16) __bf16 As[2][128 * 32];
  __shared__ __align__(16) __bf16 Bs[2][128 * 32];
  const int tid = threadIdx.x, lane = tid & 31, wave = tid >> 5;
  const int mb = blockIdx.y * 128, nb = blockIdx.x * 128;
  const int wm = (wave & 3) * 32, wn = (wave >> 2) * 64;
  v8f acc[2][4];
#pragma unroll
  for (int i = 0; i < 2; ++i)
#pragma unroll
    for (int j = 0; j < 4; ++j) acc[i][j] = vzero8();

  const int lr = tid >> 1, lk = (tid & 1) * 16;
  const __bf16* arow = Mid + (size_t)(mb + lr) * 2048 + lk;
  const float*  brow = PW + (size_t)(nb + lr) * 2048 + lk;

  {  // prologue
    const v8bf* ap = (const v8bf*)arow;
    *(v8bf*)&As[0][lr * 32 + lk] = ap[0];
    *(v8bf*)&As[0][lr * 32 + lk + 8] = ap[1];
    const float4* bp = (const float4*)brow;
    cvt_store8(&Bs[0][lr * 32 + lk], bp[0], bp[1]);
    cvt_store8(&Bs[0][lr * 32 + lk + 8], bp[2], bp[3]);
  }
  __syncthreads();

  int cur = 0;
  for (int k0 = 0; k0 < 2048; k0 += 32) {
    const bool nxt = (k0 + 32) < 2048;
    v8bf a0, a1;
    float4 b0, b1, b2, b3;
    if (nxt) {
      const v8bf* ap = (const v8bf*)(arow + k0 + 32);
      a0 = ap[0]; a1 = ap[1];
      const float4* bp = (const float4*)(brow + k0 + 32);
      b0 = bp[0]; b1 = bp[1]; b2 = bp[2]; b3 = bp[3];
    }
    v16bf af[2], bfr[4];
#pragma unroll
    for (int mi = 0; mi < 2; ++mi) af[mi] = load_afrag(As[cur], wm + mi * 16, lane);
#pragma unroll
    for (int ni = 0; ni < 4; ++ni) bfr[ni] = load_bfrag(Bs[cur], wn + ni * 16, lane);
#pragma unroll
    for (int mi = 0; mi < 2; ++mi)
#pragma unroll
      for (int ni = 0; ni < 4; ++ni) acc[mi][ni] = wmma_bf16(af[mi], bfr[ni], acc[mi][ni]);
    if (nxt) {
      *(v8bf*)&As[cur ^ 1][lr * 32 + lk] = a0;
      *(v8bf*)&As[cur ^ 1][lr * 32 + lk + 8] = a1;
      cvt_store8(&Bs[cur ^ 1][lr * 32 + lk], b0, b1);
      cvt_store8(&Bs[cur ^ 1][lr * 32 + lk + 8], b2, b3);
    }
    __syncthreads();
    cur ^= 1;
  }
  const int l = lane & 15, hi = lane >> 4;
#pragma unroll
  for (int mi = 0; mi < 2; ++mi)
#pragma unroll
    for (int ni = 0; ni < 4; ++ni) {
      const int col = nb + wn + ni * 16 + l;
      const float bc = bias[col];
#pragma unroll
      for (int r = 0; r < 8; ++r) {
        const int row = mb + wm + mi * 16 + hi * 8 + r;
        Out[(size_t)row * 1024 + col] = acc[mi][ni][r] + bc;
      }
    }
}

extern "C" void kernel_launch(void* const* d_in, const int* in_sizes, int n_in,
                              void* d_out, int out_size, void* d_ws, size_t ws_size,
                              hipStream_t stream) {
  const float* x       = (const float*)d_in[0];  // (32,512,1024)
  const float* routers = (const float*)d_in[1];  // (8,64,256)
  const float* w_kv    = (const float*)d_in[2];  // (2048,1024)
  const float* proj_w  = (const float*)d_in[3];  // (1024,2048)
  const float* proj_b  = (const float*)d_in[4];  // (1024)
  (void)in_sizes; (void)n_in; (void)out_size; (void)ws_size;

  float* out  = (float*)d_out;                     // (32,64,1024)
  float* attn = out + (size_t)32 * 64 * 1024;      // (32,8,64,512)

  __bf16* kv  = (__bf16*)d_ws;                     // (16384,2048) bf16, 64 MB
  __bf16* mid = kv + (size_t)16384 * 2048;         // (2048,2048) bf16, 8 MB

  kv_gemm_kernel<<<dim3(16, 128), 256, 0, stream>>>(x, w_kv, kv);
  score_kernel<<<256, 512, 0, stream>>>(routers, kv, attn);
  softmax_kernel<<<2048, 256, 0, stream>>>(attn);
  av_kernel<<<256, 256, 0, stream>>>(attn, kv, mid);
  proj_kernel<<<dim3(8, 16), 256, 0, stream>>>(mid, proj_w, proj_b, out);
}
